// Mamba2SSD_71562745086268
// MI455X (gfx1250) — matrix operand order repaired and checked by reading, not compiled, not run on hardware
//
#include <hip/hip_runtime.h>
#include <hip/hip_bf16.h>
#include <math.h>

typedef _Float16 half_t;
typedef __attribute__((ext_vector_type(16))) _Float16 v16h;
typedef __attribute__((ext_vector_type(8)))  float    v8f;

#define BATCH_   2
#define T_LEN    2048
#define DMODEL   1024
#define DSTATE   64
#define DTRANK   64
#define DINNER   2048
#define DPROJ    4288   // 2*DINNER + 2*DSTATE + DTRANK
#define OFF_Z    (DINNER)           // 2048
#define OFF_B    (2*DINNER)         // 4096
#define OFF_C    (2*DINNER+DSTATE)  // 4160
#define OFF_DT   (2*DINNER+2*DSTATE)// 4224

// ---------------------------------------------------------------------------
// fp32 -> fp16 conversion
// ---------------------------------------------------------------------------
__global__ void cvt_f32_to_f16(const float* __restrict__ in,
                               half_t* __restrict__ out, int n) {
  int idx = blockIdx.x * blockDim.x + threadIdx.x;
  if (idx < n) out[idx] = (half_t)in[idx];
}

// ---------------------------------------------------------------------------
// Fragment loaders (ISA 16-bit WMMA layouts, wave32)
// A 16x32: lanes 0-15 -> K 0..7 / 16..23 ; lanes 16-31 -> K 8..15 / 24..31
// B 32x16: lanes 0-15 -> K 0..7 / 16..23 ; lanes 16-31 -> K 8..15 / 24..31
// ---------------------------------------------------------------------------
__device__ __forceinline__ v16h load_afrag(const half_t* __restrict__ ap, int agrp) {
  v16h a;
#pragma unroll
  for (int e = 0; e < 8; ++e) {
    a[e]     = ap[agrp + e];
    a[8 + e] = ap[16 + agrp + e];
  }
  return a;
}
__device__ __forceinline__ v16h load_bfrag(const half_t* __restrict__ bp) {
  v16h b;
#pragma unroll
  for (int e = 0; e < 8; ++e) { b[e] = bp[e]; b[8 + e] = bp[16 + e]; }
  return b;
}

// One pipeline stage: 2 A-fragments + 4 B-fragments at K-offset kc
struct Frags { v16h a0, a1, b0, b1, b2, b3; };

__device__ __forceinline__ Frags load_stage(const half_t* __restrict__ ap0,
                                            const half_t* __restrict__ ap1,
                                            const half_t* __restrict__ bp0,
                                            size_t K, int kc, int agrp) {
  Frags f;
  f.a0 = load_afrag(ap0 + kc, agrp);
  f.a1 = load_afrag(ap1 + kc, agrp);
  f.b0 = load_bfrag(bp0 + 0 * 16 * K + kc);
  f.b1 = load_bfrag(bp0 + 1 * 16 * K + kc);
  f.b2 = load_bfrag(bp0 + 2 * 16 * K + kc);
  f.b3 = load_bfrag(bp0 + 3 * 16 * K + kc);
  return f;
}

__device__ __forceinline__ void do_wmma(v8f acc[2][4], const Frags& f) {
  acc[0][0] = __builtin_amdgcn_wmma_f32_16x16x32_f16(false, f.a0, false, f.b0, (short)0, acc[0][0], false, false);
  acc[0][1] = __builtin_amdgcn_wmma_f32_16x16x32_f16(false, f.a0, false, f.b1, (short)0, acc[0][1], false, false);
  acc[0][2] = __builtin_amdgcn_wmma_f32_16x16x32_f16(false, f.a0, false, f.b2, (short)0, acc[0][2], false, false);
  acc[0][3] = __builtin_amdgcn_wmma_f32_16x16x32_f16(false, f.a0, false, f.b3, (short)0, acc[0][3], false, false);
  acc[1][0] = __builtin_amdgcn_wmma_f32_16x16x32_f16(false, f.a1, false, f.b0, (short)0, acc[1][0], false, false);
  acc[1][1] = __builtin_amdgcn_wmma_f32_16x16x32_f16(false, f.a1, false, f.b1, (short)0, acc[1][1], false, false);
  acc[1][2] = __builtin_amdgcn_wmma_f32_16x16x32_f16(false, f.a1, false, f.b2, (short)0, acc[1][2], false, false);
  acc[1][3] = __builtin_amdgcn_wmma_f32_16x16x32_f16(false, f.a1, false, f.b3, (short)0, acc[1][3], false, false);
}

// ---------------------------------------------------------------------------
// WMMA GEMM:  C[M,N] = A[M,K] * Bw[N,K]^T   (f16 in, f32 accumulate)
// Block = 128 threads = 4 waves. Wave tile = 32(M) x 64(N):
//   2 A-frags x 4 B-frags -> 8 WMMAs per K-step (0.75 frag loads/WMMA).
// Double-buffered, 2x-unrolled K loop with peeled epilogue: each stage is a
// fresh SSA def consumed once, so the rotation needs no register copies and
// the loop body is branch-free; next-stage loads issue between WMMA batches.
// Requires M % 128 == 0, N % 64 == 0, K % 64 == 0 (true for both GEMMs) ->
// no bounds checks -> EXEC all-ones at every v_wmma (ISA requirement).
// ---------------------------------------------------------------------------
__global__ __launch_bounds__(128)
void wmma_gemm_f16f32(const half_t* __restrict__ A,
                      const half_t* __restrict__ Bw,
                      float* __restrict__ C,
                      int M, int N, int K) {
  const int lane  = threadIdx.x & 31;
  const int wave  = threadIdx.x >> 5;
  const int m0    = blockIdx.x * 128 + wave * 32;
  const int n0    = blockIdx.y * 64;
  const int l15   = lane & 15;
  const int halfw = lane >> 4;
  const int agrp  = halfw * 8;
  const int bgrp  = halfw * 8;

  const half_t* ap0 = A + (size_t)(m0 + l15) * K;        // rows m0..m0+15
  const half_t* ap1 = A + (size_t)(m0 + 16 + l15) * K;   // rows m0+16..m0+31
  const half_t* bp0 = Bw + (size_t)(n0 + l15) * K + bgrp;

  v8f acc[2][4] = {};

  Frags f0 = load_stage(ap0, ap1, bp0, (size_t)K, 0, agrp);

  int kc = 0;
  for (; kc + 64 < K; kc += 64) {
    Frags f1 = load_stage(ap0, ap1, bp0, (size_t)K, kc + 32, agrp);
    do_wmma(acc, f0);
    f0 = load_stage(ap0, ap1, bp0, (size_t)K, kc + 64, agrp);
    do_wmma(acc, f1);
  }
  // epilogue: last two K-steps (kc = K-64 already in f0)
  {
    Frags f1 = load_stage(ap0, ap1, bp0, (size_t)K, K - 32, agrp);
    do_wmma(acc, f0);
    do_wmma(acc, f1);
  }

  // C/D layout: VGPR r -> row base + r + 8*(lane>=16), col = n-tile + (lane&15)
#pragma unroll
  for (int mi = 0; mi < 2; ++mi) {
    const int rowb = m0 + mi * 16 + halfw * 8;
#pragma unroll
    for (int j = 0; j < 4; ++j)
#pragma unroll
      for (int r = 0; r < 8; ++r)
        C[(size_t)(rowb + r) * N + (n0 + j * 16 + l15)] = acc[mi][j][r];
  }
}

// ---------------------------------------------------------------------------
// Depthwise causal conv(4) + bias + SiLU  -> xc
// delta = softplus(delta_raw @ dt_proj_w^T + dt_proj_b)
// grid = (B*T, DINNER/256), block = 256 (channels)
// ---------------------------------------------------------------------------
__global__ __launch_bounds__(256)
void conv_dt_kernel(const float* __restrict__ proj,
                    const float* __restrict__ conv_w,
                    const float* __restrict__ conv_b,
                    const float* __restrict__ dt_w,
                    const float* __restrict__ dt_b,
                    float* __restrict__ xc,
                    float* __restrict__ delta) {
  __shared__ float draw[DTRANK];
  const int bt = blockIdx.x;
  const int t  = bt & (T_LEN - 1);
  const int i  = blockIdx.y * 256 + threadIdx.x;

  if (threadIdx.x < DTRANK)
    draw[threadIdx.x] = proj[(size_t)bt * DPROJ + OFF_DT + threadIdx.x];
  __syncthreads();

  // causal depthwise conv over x_inner = proj[..., :DINNER]
  float acc = conv_b[i];
#pragma unroll
  for (int k = 0; k < 4; ++k) {
    int tt = t - 3 + k;
    float xv = (tt >= 0) ? proj[(size_t)(bt - 3 + k) * DPROJ + i] : 0.f;
    acc += conv_w[i * 4 + k] * xv;
  }
  float xcv = acc / (1.f + expf(-acc));           // SiLU
  xc[(size_t)bt * DINNER + i] = xcv;

  float d = dt_b[i];
#pragma unroll
  for (int r = 0; r < DTRANK; ++r) d += draw[r] * dt_w[i * DTRANK + r];
  delta[(size_t)bt * DINNER + i] = (d > 20.f) ? d : log1pf(expf(d)); // softplus
}

// ---------------------------------------------------------------------------
// Selective scan. Thread = (batch b, channel i); h[64] lives in VGPRs for the
// entire T loop. B_t / C_t double-buffered in LDS, one barrier per step.
// Emits g = (y + D*xc) * silu(z) as f16 for the out-proj GEMM, and h_T.
// grid = BATCH * (DINNER/256), block = 256
// ---------------------------------------------------------------------------
__global__ __launch_bounds__(256)
void scan_kernel(const float* __restrict__ proj,
                 const float* __restrict__ xc,
                 const float* __restrict__ delta,
                 const float* __restrict__ A_log,
                 const float* __restrict__ Dp,
                 half_t* __restrict__ g,
                 float* __restrict__ hT_out) {
  __shared__ float a2[DSTATE];
  __shared__ float Bb[2][DSTATE];
  __shared__ float Cb[2][DSTATE];

  const int b = blockIdx.x >> 3;
  const int i = (blockIdx.x & 7) * 256 + threadIdx.x;

  if (threadIdx.x < DSTATE)  // A = -exp(A_log); fold in log2(e) for exp2
    a2[threadIdx.x] = -expf(A_log[threadIdx.x]) * 1.44269504088896340736f;

  const float dpar = Dp[i];
  float h[DSTATE];
#pragma unroll
  for (int s = 0; s < DSTATE; ++s) h[s] = 0.f;

  for (int t = 0; t < T_LEN; ++t) {
    const size_t bt = (size_t)b * T_LEN + t;
    const int p = t & 1;
    if (threadIdx.x < DSTATE) {
      Bb[p][threadIdx.x] = proj[bt * DPROJ + OFF_B + threadIdx.x];
      // pull the B row for t+8 toward the caches (global_prefetch_b8)
      if (t + 8 < T_LEN)
        __builtin_prefetch(&proj[(bt + 8) * DPROJ + OFF_B + threadIdx.x], 0, 0);
    } else if (threadIdx.x < 2 * DSTATE) {
      Cb[p][threadIdx.x - DSTATE] = proj[bt * DPROJ + OFF_C + (threadIdx.x - DSTATE)];
    }
    __syncthreads();

    const float dt = delta[bt * DINNER + i];
    const float xv = xc[bt * DINNER + i];
    const float dx = dt * xv;
    float y = 0.f;
#pragma unroll
    for (int s = 0; s < DSTATE; ++s) {
      float hn = exp2f(a2[s] * dt) * h[s] + dx * Bb[p][s];
      h[s] = hn;
      y += hn * Cb[p][s];
    }
    const float zv = proj[bt * DPROJ + OFF_Z + i];
    const float yv = (y + dpar * xv) * (zv / (1.f + expf(-zv)));
    g[bt * DINNER + i] = (half_t)yv;
  }

#pragma unroll
  for (int s = 0; s < DSTATE; ++s)
    hT_out[((size_t)b * DINNER + i) * DSTATE + s] = h[s];
}

// ---------------------------------------------------------------------------
extern "C" void kernel_launch(void* const* d_in, const int* in_sizes, int n_in,
                              void* d_out, int out_size, void* d_ws, size_t ws_size,
                              hipStream_t stream) {
  (void)in_sizes; (void)n_in; (void)out_size; (void)ws_size;

  const float* x        = (const float*)d_in[0]; // (B,T,DMODEL)
  const float* in_w     = (const float*)d_in[1]; // (DPROJ,DMODEL)
  const float* conv_w   = (const float*)d_in[2]; // (DINNER,1,4)
  const float* conv_b   = (const float*)d_in[3]; // (DINNER)
  const float* A_log    = (const float*)d_in[4]; // (DSTATE)
  const float* D_param  = (const float*)d_in[5]; // (DINNER)
  const float* dt_w     = (const float*)d_in[6]; // (DINNER,DTRANK)
  const float* dt_b     = (const float*)d_in[7]; // (DINNER)
  const float* out_w    = (const float*)d_in[8]; // (DMODEL,DINNER)

  float* out = (float*)d_out;                       // (B,T,DMODEL) then hT
  float* hT  = out + (size_t)BATCH_ * T_LEN * DMODEL;

  const int BT = BATCH_ * T_LEN;                    // 4096

  // workspace layout
  char* ws = (char*)d_ws;
  half_t* x16    = (half_t*)ws;                                ws += (size_t)BT * DMODEL * 2;
  half_t* winp16 = (half_t*)ws;                                ws += (size_t)DPROJ * DMODEL * 2;
  half_t* wout16 = (half_t*)ws;                                ws += (size_t)DMODEL * DINNER * 2;
  float*  proj   = (float*)ws;                                 ws += (size_t)BT * DPROJ * 4;
  float*  xc     = (float*)ws;                                 ws += (size_t)BT * DINNER * 4;
  float*  delta  = (float*)ws;                                 ws += (size_t)BT * DINNER * 4;
  half_t* g16    = (half_t*)ws;                                ws += (size_t)BT * DINNER * 2;

  // 1) fp32 -> fp16 for WMMA operands
  {
    int n = BT * DMODEL;
    cvt_f32_to_f16<<<(n + 255) / 256, 256, 0, stream>>>(x, x16, n);
    n = DPROJ * DMODEL;
    cvt_f32_to_f16<<<(n + 255) / 256, 256, 0, stream>>>(in_w, winp16, n);
    n = DMODEL * DINNER;
    cvt_f32_to_f16<<<(n + 255) / 256, 256, 0, stream>>>(out_w, wout16, n);
  }

  // 2) in_proj GEMM: proj[BT,DPROJ] = x16[BT,DMODEL] * winp16[DPROJ,DMODEL]^T
  {
    dim3 grid(BT / 128, DPROJ / 64);   // (32, 67)
    wmma_gemm_f16f32<<<grid, 128, 0, stream>>>(x16, winp16, proj, BT, DPROJ, DMODEL);
  }

  // 3) depthwise conv + SiLU, dt projection + softplus
  {
    dim3 grid(BT, DINNER / 256);
    conv_dt_kernel<<<grid, 256, 0, stream>>>(proj, conv_w, conv_b, dt_w, dt_b, xc, delta);
  }

  // 4) sequential selective scan (state in VGPRs), emits gated f16 activations + hT
  {
    dim3 grid(BATCH_ * (DINNER / 256));  // 16 blocks
    scan_kernel<<<grid, 256, 0, stream>>>(proj, xc, delta, A_log, D_param, g16, hT);
  }

  // 5) out_proj GEMM: out[BT,DMODEL] = g16[BT,DINNER] * wout16[DMODEL,DINNER]^T
  {
    dim3 grid(BT / 128, DMODEL / 64);  // (32, 16)
    wmma_gemm_f16f32<<<grid, 128, 0, stream>>>(g16, wout16, out, BT, DMODEL, DINNER);
  }
}
